// GPT2MultiHeadAttention_11974368821414
// MI455X (gfx1250) — matrix operand order repaired and checked by reading, not compiled, not run on hardware
//
#include <hip/hip_runtime.h>
#include <cstdint>
#include <cstddef>

// Problem constants (GPT-2 MHA: B=2, S=2048, E=1024, H=16, D=64)
#define B_  2
#define S_  2048
#define E_  1024
#define H_  16
#define D_  64
#define BS_ (B_ * S_)   // 4096 total rows

typedef __attribute__((ext_vector_type(16))) __bf16 v16bf;
typedef __attribute__((ext_vector_type(8)))  float  v8f;
typedef __attribute__((ext_vector_type(4)))  int    i32x4;

struct U32x8 { uint4 lo; uint4 hi; };

// --- CDNA5 async global->LDS path (guarded; fallback = synchronous staging) ---
#if defined(__has_builtin)
#if __has_builtin(__builtin_amdgcn_global_load_async_to_lds_b128) && \
    __has_builtin(__builtin_amdgcn_s_wait_asynccnt)
#define HAVE_ASYNC_LDS 1
#endif
#endif
#ifndef HAVE_ASYNC_LDS
#define HAVE_ASYNC_LDS 0
#endif

#if HAVE_ASYNC_LDS
#define ASYNC_WAIT(n) __builtin_amdgcn_s_wait_asynccnt(n)
typedef __attribute__((address_space(1))) i32x4* gvec_p;
typedef __attribute__((address_space(3))) i32x4* lvec_p;
#else
#define ASYNC_WAIT(n)
#endif

static __device__ __forceinline__ unsigned short f2bf(float f) {
  // round-to-nearest-even f32 -> bf16
  unsigned int u = __float_as_uint(f);
  unsigned int r = (u + 0x7FFFu + ((u >> 16) & 1u)) >> 16;
  return (unsigned short)r;
}

static __device__ __forceinline__ v8f wmma_bf16(v16bf a, v16bf b, v8f c) {
  // D = A(16x32 bf16) * B(32x16 bf16) + C(16x16 f32)
  return __builtin_amdgcn_wmma_f32_16x16x32_bf16(
      /*neg_a=*/false, a, /*neg_b=*/false, b,
      /*c_mod=*/(short)0, c, /*reuse_a=*/false, /*reuse_b=*/false);
}

// A-operand fragment: 16x32 bf16 from row-major [rows, ld] at (row0, k0).
// ISA layout: lanes 0-15 hold K = 0..7 & 16..23 of row (lane&15);
//             lanes 16-31 hold K = 8..15 & 24..31.
static __device__ __forceinline__ v16bf load_afrag(
    const unsigned short* base, int ld, int row0, int k0) {
  int lane = (int)(threadIdx.x & 31u);
  const unsigned short* p = base + (size_t)(row0 + (lane & 15)) * ld
                                 + k0 + ((lane >> 4) << 3);
  U32x8 pr;
  pr.lo = *(const uint4*)(p);       // K = ks .. ks+7
  pr.hi = *(const uint4*)(p + 16);  // K = 16+ks .. 16+ks+7
  return __builtin_bit_cast(v16bf, pr);
}

// B-operand fragment: 32x16 bf16 from N-major [N, ld] (i.e. B^T row-major).
// ISA layout: lanes 0-15 hold K = 0..7 & 16..23 of column (lane&15);
//             lanes 16-31 hold K = 8..15 & 24..31.
static __device__ __forceinline__ v16bf load_bfrag(
    const unsigned short* base, int ld, int n0, int k0) {
  int lane = (int)(threadIdx.x & 31u);
  const unsigned short* p = base + (size_t)(n0 + (lane & 15)) * ld
                                 + k0 + ((lane >> 4) << 3);
  U32x8 pr;
  pr.lo = *(const uint4*)(p);
  pr.hi = *(const uint4*)(p + 16);
  return __builtin_bit_cast(v16bf, pr);
}

// Cooperatively stage a 64(N) x 32(K) bf16 B-tile (4 KB) into LDS.
// 128 threads, 2 x b128 chunks each. Async path: 2 ASYNCcnt per wave per call.
static __device__ __forceinline__ void stage_b_tile(
    const unsigned short* __restrict__ gsrc,  // points at (row n0, col k)
    int ld, unsigned short* lbuf) {           // lbuf: [64][32] bf16
  int t = (int)threadIdx.x;  // 0..127
#pragma unroll
  for (int c = t; c < 256; c += 128) {
    int row = c >> 2;
    int ke  = (c & 3) * 8;
#if HAVE_ASYNC_LDS
    __builtin_amdgcn_global_load_async_to_lds_b128(
        (gvec_p)(gsrc + (size_t)row * ld + ke),
        (lvec_p)(lbuf + row * 32 + ke),
        0, 0);
#else
    *(uint4*)(lbuf + row * 32 + ke) = *(const uint4*)(gsrc + (size_t)row * ld + ke);
#endif
  }
}

// ---------------- pre-pass casts ----------------

__global__ void k_cast(const float* __restrict__ in,
                       unsigned short* __restrict__ out, int n) {
  int i = blockIdx.x * blockDim.x + threadIdx.x;
  if (i < n) out[i] = f2bf(in[i]);
}

// in: [H][K][N] f32  ->  out: [H][N][K] bf16 (transpose-cast, output-linear)
__global__ void k_tcast(const float* __restrict__ in,
                        unsigned short* __restrict__ out,
                        int Hn, int Kn, int Nn) {
  int i = blockIdx.x * blockDim.x + threadIdx.x;
  int tot = Hn * Kn * Nn;
  if (i >= tot) return;
  int k = i % Kn;
  int t = i / Kn;
  int n = t % Nn;
  int h = t / Nn;
  out[i] = f2bf(in[((size_t)h * Kn + k) * Nn + n]);
}

// ---------------- QKV projection GEMM ----------------
// per head: X[4096,1024] * W (as B-operand, staged in LDS) -> [4096,64] (+bias)
// grid = (32, H), block = 128 (4 waves; wave computes 32 rows x 64 cols)
__global__ void __launch_bounds__(128) k_qkv(
    const unsigned short* __restrict__ Xbf,   // [BS][E] bf16
    const unsigned short* __restrict__ Wt,    // [H][D][E] bf16 (N-major)
    const float* __restrict__ bias,           // [H][D]
    unsigned short* __restrict__ out,         // [B][H][S][D] or [B][H][D][S]
    int transposed) {
  __shared__ unsigned short bst[2][64 * 32];  // double-buffered B tile (8 KB)

  int wave = (int)(threadIdx.x >> 5);
  int lane = (int)(threadIdx.x & 31u);
  int h    = (int)blockIdx.y;
  int g0   = (int)blockIdx.x * 128 + wave * 32;   // first of 32 rows
  int b    = g0 >> 11;                            // S_ = 2048, 32 | 2048
  int s0   = g0 & (S_ - 1);

  const unsigned short* Bg = Wt + (size_t)h * D_ * E_;

  v8f acc[2][4] = {};
  stage_b_tile(Bg, E_, &bst[0][0]);               // prologue: K-step 0

  const int NK = E_ / 32;                         // 32 K-steps
  for (int i = 0; i < NK; i++) {
    int k = i * 32;
    __syncthreads();                              // prior compute done -> safe overwrite
    if (i + 1 < NK) {
      stage_b_tile(Bg + (i + 1) * 32, E_, &bst[(i + 1) & 1][0]);
      ASYNC_WAIT(2);                              // this wave's step-i copies landed
    } else {
      ASYNC_WAIT(0);
    }
    __syncthreads();                              // publish step-i tile block-wide

    v16bf a0 = load_afrag(Xbf, E_, g0, k);
    v16bf a1 = load_afrag(Xbf, E_, g0 + 16, k);
    const unsigned short* lb = &bst[i & 1][0];
#pragma unroll
    for (int nt = 0; nt < 4; nt++) {
      v16bf bf = load_bfrag(lb, 32, nt * 16, 0);
      acc[0][nt] = wmma_bf16(a0, bf, acc[0][nt]);
      acc[1][nt] = wmma_bf16(a1, bf, acc[1][nt]);
    }
  }

  int col = lane & 15;
  int rb  = (lane >> 4) * 8;
#pragma unroll
  for (int mt = 0; mt < 2; mt++) {
    int sm = s0 + mt * 16;
    if (!transposed) {
      // Q/K: [B][H][S][D]
#pragma unroll
      for (int nt = 0; nt < 4; nt++) {
        int d = nt * 16 + col;
        float bv = bias[h * D_ + d];
        size_t base = (((size_t)b * H_ + h) * S_ + sm) * D_ + d;
#pragma unroll
        for (int r = 0; r < 8; r++)
          out[base + (size_t)(rb + r) * D_] = f2bf(acc[mt][nt][r] + bv);
      }
    } else {
      // V: [B][H][D][S] (transposed so it is the B-operand of P*V)
#pragma unroll
      for (int nt = 0; nt < 4; nt++) {
        int d = nt * 16 + col;
        float bv = bias[h * D_ + d];
        unsigned int pk[4];
#pragma unroll
        for (int r = 0; r < 4; r++) {
          unsigned int lo = f2bf(acc[mt][nt][2 * r] + bv);
          unsigned int hi = f2bf(acc[mt][nt][2 * r + 1] + bv);
          pk[r] = lo | (hi << 16);
        }
        size_t base = (((size_t)b * H_ + h) * D_ + d) * S_ + sm + rb;
        *(uint4*)(out + base) = make_uint4(pk[0], pk[1], pk[2], pk[3]);
      }
    }
  }
}

// ---------------- causal flash attention ----------------
// grid = (S/64, B*H), block = 128 (4 waves; wave owns 16 query rows)
__global__ void __launch_bounds__(128) k_attn(
    const unsigned short* __restrict__ Qb,  // [B][H][S][D]
    const unsigned short* __restrict__ Kb,  // [B][H][S][D]
    const unsigned short* __restrict__ Vt,  // [B][H][D][S]
    unsigned short* __restrict__ Ctx) {     // [BS][E] bf16
  __shared__ unsigned short lds[4][16 * 32];

  int wave = (int)(threadIdx.x >> 5);
  int lane = (int)(threadIdx.x & 31u);
  int bh   = (int)blockIdx.y;
  int b    = bh >> 4;
  int h    = bh & 15;
  int s0   = ((int)blockIdx.x * 4 + wave) * 16;

  const unsigned short* Q = Qb + (size_t)bh * S_ * D_;
  const unsigned short* K = Kb + (size_t)bh * S_ * D_;
  const unsigned short* V = Vt + (size_t)bh * D_ * S_;

  v16bf q0 = load_afrag(Q, D_, s0, 0);
  v16bf q1 = load_afrag(Q, D_, s0, 32);

  v8f acc[4] = {};
  float mi[8], li[8];
#pragma unroll
  for (int r = 0; r < 8; r++) { mi[r] = -1e30f; li[r] = 0.0f; }

  int col = lane & 15;
  int rb  = (lane >> 4) * 8;
  const float scale = 0.125f;  // 1/sqrt(64)

  for (int t0 = 0; t0 <= s0 + 15; t0 += 32) {
    // scores: two 16x16 tiles (t columns t0..t0+15 and t0+16..t0+31)
    v8f sc[2];
#pragma unroll
    for (int f = 0; f < 2; f++) {
      v8f z = {};
      v16bf k0 = load_bfrag(K, D_, t0 + 16 * f, 0);
      v16bf k1 = load_bfrag(K, D_, t0 + 16 * f, 32);
      z = wmma_bf16(q0, k0, z);
      z = wmma_bf16(q1, k1, z);
      sc[f] = z;
    }
    // causal mask + scale
#pragma unroll
    for (int f = 0; f < 2; f++) {
      int t = t0 + 16 * f + col;
#pragma unroll
      for (int r = 0; r < 8; r++) {
        int s = s0 + rb + r;
        float v = sc[f][r] * scale;
        sc[f][r] = (t <= s) ? v : -1e30f;
      }
    }
    // online softmax: row max over 16 lanes of each half-wave
    float rmax[8];
#pragma unroll
    for (int r = 0; r < 8; r++) rmax[r] = fmaxf(sc[0][r], sc[1][r]);
#pragma unroll
    for (int m = 1; m < 16; m <<= 1)
#pragma unroll
      for (int r = 0; r < 8; r++)
        rmax[r] = fmaxf(rmax[r], __shfl_xor(rmax[r], m, 32));

    float corr[8];
#pragma unroll
    for (int r = 0; r < 8; r++) {
      float mn = fmaxf(mi[r], rmax[r]);
      corr[r] = __expf(mi[r] - mn);
      mi[r] = mn;
      li[r] *= corr[r];
    }
    float p[2][8];
#pragma unroll
    for (int f = 0; f < 2; f++)
#pragma unroll
      for (int r = 0; r < 8; r++) p[f][r] = __expf(sc[f][r] - mi[r]);
#pragma unroll
    for (int r = 0; r < 8; r++) li[r] += p[0][r] + p[1][r];
#pragma unroll
    for (int nt = 0; nt < 4; nt++)
#pragma unroll
      for (int r = 0; r < 8; r++) acc[nt][r] *= corr[r];

    // P tile (C-layout f32) -> LDS -> A-operand bf16 fragment
    unsigned short* pl = &lds[wave][0];
#pragma unroll
    for (int f = 0; f < 2; f++)
#pragma unroll
      for (int r = 0; r < 8; r++)
        pl[(rb + r) * 32 + 16 * f + col] = f2bf(p[f][r]);
    v16bf pf = load_afrag(pl, 32, 0, 0);

    // out += P(16x32) * V-chunk(32x64)
#pragma unroll
    for (int nt = 0; nt < 4; nt++) {
      v16bf vf = load_bfrag(V, S_, nt * 16, t0);
      acc[nt] = wmma_bf16(pf, vf, acc[nt]);
    }
  }

  // finish softmax normalization: reduce per-column partial sums
#pragma unroll
  for (int m = 1; m < 16; m <<= 1)
#pragma unroll
    for (int r = 0; r < 8; r++) li[r] += __shfl_xor(li[r], m, 32);
  float inv[8];
#pragma unroll
  for (int r = 0; r < 8; r++) inv[r] = 1.0f / li[r];

  // store context (heads concatenated): Ctx[b*S+s][h*64+d]
  size_t row0 = (size_t)b * S_ + s0 + rb;
#pragma unroll
  for (int nt = 0; nt < 4; nt++) {
    int e = h * 64 + nt * 16 + col;
#pragma unroll
    for (int r = 0; r < 8; r++)
      Ctx[(row0 + r) * E_ + e] = f2bf(acc[nt][r] * inv[r]);
  }
}

// ---------------- output projection ----------------
// Out[4096,1024] = Ctx * Wproj + b_proj
// grid = (32, 16), block = 128 (wave computes 32 rows x 64 cols; B in LDS)
__global__ void __launch_bounds__(128) k_proj(
    const unsigned short* __restrict__ Ctx,  // [BS][E] bf16
    const unsigned short* __restrict__ WpT,  // [E][E] bf16 (N-major)
    const float* __restrict__ bproj,         // [E]
    float* __restrict__ Out) {               // [BS][E] f32
  __shared__ unsigned short bst[2][64 * 32];

  int wave = (int)(threadIdx.x >> 5);
  int lane = (int)(threadIdx.x & 31u);
  int m0 = (int)blockIdx.x * 128 + wave * 32;
  int n0 = (int)blockIdx.y * 64;

  const unsigned short* Bg = WpT + (size_t)n0 * E_;

  v8f acc[2][4] = {};
  stage_b_tile(Bg, E_, &bst[0][0]);

  const int NK = E_ / 32;
  for (int i = 0; i < NK; i++) {
    int k = i * 32;
    __syncthreads();
    if (i + 1 < NK) {
      stage_b_tile(Bg + (i + 1) * 32, E_, &bst[(i + 1) & 1][0]);
      ASYNC_WAIT(2);
    } else {
      ASYNC_WAIT(0);
    }
    __syncthreads();

    v16bf a0 = load_afrag(Ctx, E_, m0, k);
    v16bf a1 = load_afrag(Ctx, E_, m0 + 16, k);
    const unsigned short* lb = &bst[i & 1][0];
#pragma unroll
    for (int nt = 0; nt < 4; nt++) {
      v16bf bf = load_bfrag(lb, 32, nt * 16, 0);
      acc[0][nt] = wmma_bf16(a0, bf, acc[0][nt]);
      acc[1][nt] = wmma_bf16(a1, bf, acc[1][nt]);
    }
  }

  int col = lane & 15;
  int rb  = (lane >> 4) * 8;
#pragma unroll
  for (int mt = 0; mt < 2; mt++) {
#pragma unroll
    for (int nt = 0; nt < 4; nt++) {
      int n = n0 + nt * 16 + col;
      float bb = bproj[n];
#pragma unroll
      for (int r = 0; r < 8; r++)
        Out[(size_t)(m0 + mt * 16 + rb + r) * E_ + n] = acc[mt][nt][r] + bb;
    }
  }
}

// ---------------- host-side launch ----------------

extern "C" void kernel_launch(void* const* d_in, const int* in_sizes, int n_in,
                              void* d_out, int out_size, void* d_ws, size_t ws_size,
                              hipStream_t stream) {
  const float* x  = (const float*)d_in[0];
  const float* wq = (const float*)d_in[1];
  const float* bq = (const float*)d_in[2];
  const float* wk = (const float*)d_in[3];
  const float* bk = (const float*)d_in[4];
  const float* wv = (const float*)d_in[5];
  const float* bv = (const float*)d_in[6];
  const float* wp = (const float*)d_in[7];
  const float* bp = (const float*)d_in[8];
  float* out = (float*)d_out;

  uint8_t* w = (uint8_t*)d_ws;
  unsigned short* Xbf = (unsigned short*)w; w += (size_t)BS_ * E_ * 2;           // 8 MB
  unsigned short* WqT = (unsigned short*)w; w += (size_t)H_ * D_ * E_ * 2;       // 2 MB
  unsigned short* WkT = (unsigned short*)w; w += (size_t)H_ * D_ * E_ * 2;       // 2 MB
  unsigned short* WvT = (unsigned short*)w; w += (size_t)H_ * D_ * E_ * 2;       // 2 MB
  unsigned short* WpT = (unsigned short*)w; w += (size_t)E_ * E_ * 2;            // 2 MB
  unsigned short* Qb  = (unsigned short*)w; w += (size_t)B_ * H_ * S_ * D_ * 2;  // 8 MB
  unsigned short* Kb  = (unsigned short*)w; w += (size_t)B_ * H_ * S_ * D_ * 2;  // 8 MB
  unsigned short* Vtw = (unsigned short*)w; w += (size_t)B_ * H_ * D_ * S_ * 2;  // 8 MB
  unsigned short* Ctx = (unsigned short*)w; w += (size_t)BS_ * E_ * 2;           // 8 MB

  // 1) casts / transposes to bf16
  k_cast<<<(BS_ * E_ + 255) / 256, 256, 0, stream>>>(x, Xbf, BS_ * E_);
  k_tcast<<<(H_ * E_ * D_ + 255) / 256, 256, 0, stream>>>(wq, WqT, H_, E_, D_);
  k_tcast<<<(H_ * E_ * D_ + 255) / 256, 256, 0, stream>>>(wk, WkT, H_, E_, D_);
  k_tcast<<<(H_ * E_ * D_ + 255) / 256, 256, 0, stream>>>(wv, WvT, H_, E_, D_);
  k_tcast<<<(E_ * E_ + 255) / 256, 256, 0, stream>>>(wp, WpT, 1, E_, E_);

  // 2) QKV projections (WMMA GEMMs, async-LDS staged B)
  dim3 gq(BS_ / 128, H_);  // (32, 16)
  k_qkv<<<gq, 128, 0, stream>>>(Xbf, WqT, bq, Qb, 0);
  k_qkv<<<gq, 128, 0, stream>>>(Xbf, WkT, bk, Kb, 0);
  k_qkv<<<gq, 128, 0, stream>>>(Xbf, WvT, bv, Vtw, 1);

  // 3) causal flash attention (WMMA + online softmax)
  dim3 ga(S_ / 64, B_ * H_);  // (32, 32)
  k_attn<<<ga, 128, 0, stream>>>(Qb, Kb, Vtw, Ctx);

  // 4) output projection (WMMA GEMM, async-LDS staged B, f32 out + bias)
  dim3 gp(BS_ / 128, E_ / 64);  // (32, 16)
  k_proj<<<gp, 128, 0, stream>>>(Ctx, WpT, bp, out);
}
